// SGSEncoder_4183298146979
// MI455X (gfx1250) — matrix operand order repaired and checked by reading, not compiled, not run on hardware
//
#include <hip/hip_runtime.h>
#include <math.h>

typedef __bf16 bf16_t;
typedef __attribute__((ext_vector_type(16))) __bf16 v16bf;
typedef __attribute__((ext_vector_type(8)))  float  v8f;

#define B_   256
#define S_   512
#define DS_  64
#define DF_  300
#define NTOK (B_*S_)
#define CDIM 600
#define KPAD 608
#define N2P  320          // FFN2 output dim padded 300 -> 320 (20 tiles)
#define LDST 616          // LDS A-panel row stride (bank-conflict-free)
#define LN_EPS_ 1e-5f

__device__ __forceinline__ unsigned short f2bf_bits(float f) {
  unsigned u = __builtin_bit_cast(unsigned, f);
  unsigned r = (u + 0x7FFFu + ((u >> 16) & 1u)) >> 16;
  return (unsigned short)r;
}
__device__ __forceinline__ bf16_t f2bf(float f) {
  unsigned short h = f2bf_bits(f);
  return __builtin_bit_cast(bf16_t, h);
}
__device__ __forceinline__ float sigm(float x) { return 1.f / (1.f + expf(-x)); }

// ---------------------------------------------------------------- weight cvt
__global__ void convert_w_kernel(const float* __restrict__ src, bf16_t* __restrict__ dst,
                                 int npass, int rows_d, int cols_d, int rows_s, int cols_s) {
  long total = (long)npass * rows_d * cols_d;
  for (long i = (long)blockIdx.x * blockDim.x + threadIdx.x; i < total;
       i += (long)gridDim.x * blockDim.x) {
    int c = (int)(i % cols_d);
    long t = i / cols_d;
    int r = (int)(t % rows_d);
    int p = (int)(t / rows_d);
    float v = (r < rows_s && c < cols_s) ? src[((long)p * rows_s + r) * cols_s + c] : 0.f;
    dst[i] = f2bf(v);
  }
}

// ---------------------------------------------------------------- gather/prep
__global__ void __launch_bounds__(256) prep_kernel(
    const int* __restrict__ ids, const unsigned char* __restrict__ mask,
    const float* __restrict__ mu_table, const float* __restrict__ logvar_table,
    const float* __restrict__ alpha_table, const float* __restrict__ feat_table,
    const float* __restrict__ pos_mu, const float* __restrict__ pos_alpha,
    float* __restrict__ mu, float* __restrict__ inv_var,
    float* __restrict__ alpha, float* __restrict__ features) {
  int tok = blockIdx.x;
  int s = tok & (S_ - 1);
  int id = ids[tok];
  int t = threadIdx.x;
  if (t < DS_) {
    mu[(long)tok * DS_ + t]      = mu_table[(long)id * DS_ + t] + pos_mu[s * DS_ + t];
    inv_var[(long)tok * DS_ + t] = expf(-logvar_table[(long)id * DS_ + t]);
  }
  for (int f = t; f < DF_; f += 256)
    features[(long)tok * DF_ + f] = feat_table[(long)id * DF_ + f];
  if (t == 0) {
    float m = mask[tok] ? 1.f : 0.f;
    alpha[tok] = sigm(alpha_table[id]) * sigm(pos_alpha[s]) * m;
  }
}

// ---------------------------------------------------------------- render
__global__ void __launch_bounds__(256) render_kernel(
    const float* __restrict__ mu, const float* __restrict__ inv_var,
    const float* __restrict__ alpha, const float* __restrict__ features,
    const unsigned char* __restrict__ mask, const float* __restrict__ log_tau,
    float* __restrict__ meaning) {
  __shared__ float cen[DS_];
  __shared__ float wls[S_];
  __shared__ float red[256];
  int b = blockIdx.x, t = threadIdx.x;
  const int base = b * S_;
  if (t < DS_) {
    float acc = 0.f, len = 0.f;
    for (int s = 0; s < S_; ++s) {
      float m = mask[base + s] ? 1.f : 0.f;
      acc += mu[(long)(base + s) * DS_ + t] * m;
      len += m;
    }
    cen[t] = acc / fmaxf(len, 1.f);
  }
  __syncthreads();
  float inv2tau = -0.5f / expf(log_tau[0]);
  float part = 0.f;
  for (int s = t; s < S_; s += 256) {
    const float* mp = mu + (long)(base + s) * DS_;
    const float* ip = inv_var + (long)(base + s) * DS_;
    float d2 = 0.f;
    for (int d = 0; d < DS_; ++d) { float df = cen[d] - mp[d]; d2 += df * df * ip[d]; }
    float m = mask[base + s] ? 1.f : 0.f;
    float w = alpha[base + s] * expf(d2 * inv2tau) * m;
    wls[s] = w;
    part += w;
  }
  red[t] = part;
  __syncthreads();
  for (int o = 128; o; o >>= 1) {
    if (t < o) red[t] += red[t + o];
    __syncthreads();
  }
  float inv = 1.f / (red[0] + 1e-8f);
  for (int f = t; f < DF_; f += 256) {
    float acc = 0.f;
    for (int s = 0; s < S_; ++s) acc += wls[s] * features[(long)(base + s) * DF_ + f];
    meaning[(long)b * DF_ + f] = acc * inv;
  }
}

// ------------------------------------------------ LN stats + gate (wave/token)
__global__ void __launch_bounds__(256) stats_kernel(
    const float* __restrict__ features, const float* __restrict__ meaning,
    const float* __restrict__ gate_W, const float* __restrict__ gate_b,
    const unsigned char* __restrict__ mask, float* __restrict__ alpha,
    float* __restrict__ stats, int p) {
  int tok = blockIdx.x * 8 + (threadIdx.x >> 5);
  int lane = threadIdx.x & 31;
  int b = tok >> 9;  // /S_
  const float* gw = gate_W + p * CDIM;
  float sum = 0.f, sq = 0.f, gd = 0.f;
  for (int c = lane; c < CDIM; c += 32) {
    float x = (c < DF_) ? features[(long)tok * DF_ + c] : meaning[(long)b * DF_ + c - DF_];
    sum += x; sq += x * x; gd += x * gw[c];
  }
  for (int o = 16; o; o >>= 1) {
    sum += __shfl_xor(sum, o, 32);
    sq  += __shfl_xor(sq,  o, 32);
    gd  += __shfl_xor(gd,  o, 32);
  }
  if (lane == 0) {
    float mean = sum / (float)CDIM;
    float var  = sq / (float)CDIM - mean * mean;
    stats[(long)tok * 2]     = mean;
    stats[(long)tok * 2 + 1] = rsqrtf(var + LN_EPS_);
    float m = mask[tok] ? 1.f : 0.f;
    alpha[tok] *= sigm(gd + gate_b[p]) * m;
  }
}

// ---------------------------------------------------------------- WMMA helpers
__device__ __forceinline__ v16bf load_bf16x16(const bf16_t* p0, const bf16_t* p1) {
  v16bf v;
  ((uint4*)&v)[0] = *(const uint4*)p0;
  ((uint4*)&v)[1] = *(const uint4*)p1;
  return v;
}
__device__ __forceinline__ v8f wmma_bf16(v16bf a, v16bf b, v8f c) {
  return __builtin_amdgcn_wmma_f32_16x16x32_bf16(false, a, false, b, (short)0, c, false, false);
}

// ---------------------------------------------------------------- mu update GEMM
__global__ void __launch_bounds__(256) mu_gemm_kernel(
    const float* __restrict__ features, const float* __restrict__ meaning,
    const bf16_t* __restrict__ muWbf, const float* __restrict__ mu_b,
    float* __restrict__ mu, int p) {
  __shared__ __align__(16) bf16_t panel[128 * LDST];
  int t = threadIdx.x;
  int tok0 = blockIdx.x * 128;

  // Stage combined = [features | meaning] as bf16, pair-packed b32 stores.
  // Pairs never straddle the 300/600 boundaries (both even).
  for (int i = t; i < 128 * (KPAD / 2); i += 256) {
    int r = i / (KPAD / 2);
    int c = (i - r * (KPAD / 2)) * 2;
    int tok = tok0 + r;
    float2 x;
    if (c < DF_)       x = *(const float2*)(features + (long)tok * DF_ + c);
    else if (c < CDIM) x = *(const float2*)(meaning + (long)(tok >> 9) * DF_ + c - DF_);
    else               x = make_float2(0.f, 0.f);
    unsigned pk = (unsigned)f2bf_bits(x.x) | ((unsigned)f2bf_bits(x.y) << 16);
    ((unsigned*)panel)[(r * LDST + c) >> 1] = pk;
  }
  __syncthreads();

  int wave = t >> 5, lane = t & 31;
  int col = lane & 15;
  int kh = (lane < 16) ? 0 : 16;
  int ksel = (lane < 16) ? 0 : 16;
  const bf16_t* abase = panel + (wave * 16 + (lane & 15)) * LDST;
  const bf16_t* W = muWbf + (long)p * DS_ * KPAD;
  int tokenM = tok0 + wave * 16;

  v8f acc0 = {}, acc1 = {}, acc2 = {}, acc3 = {};
#pragma unroll 1
  for (int k0 = 0; k0 < KPAD; k0 += 32) {
    v16bf a = load_bf16x16(abase + k0 + ksel, abase + k0 + ksel + 8);
    const bf16_t* wk = W + k0 + kh;
    v16bf b0 = load_bf16x16(wk + (long)(0 * 16 + col) * KPAD, wk + (long)(0 * 16 + col) * KPAD + 8);
    v16bf b1 = load_bf16x16(wk + (long)(1 * 16 + col) * KPAD, wk + (long)(1 * 16 + col) * KPAD + 8);
    v16bf b2 = load_bf16x16(wk + (long)(2 * 16 + col) * KPAD, wk + (long)(2 * 16 + col) * KPAD + 8);
    v16bf b3 = load_bf16x16(wk + (long)(3 * 16 + col) * KPAD, wk + (long)(3 * 16 + col) * KPAD + 8);
    acc0 = wmma_bf16(a, b0, acc0);
    acc1 = wmma_bf16(a, b1, acc1);
    acc2 = wmma_bf16(a, b2, acc2);
    acc3 = wmma_bf16(a, b3, acc3);
  }
  int rowoff = (lane < 16) ? 0 : 8;
#pragma unroll
  for (int v = 0; v < 8; ++v) {
    long r = (long)(tokenM + rowoff + v) * DS_;
    int d0 = col, d1 = 16 + col, d2 = 32 + col, d3 = 48 + col;
    mu[r + d0] += tanhf(acc0[v] + mu_b[p * DS_ + d0]);
    mu[r + d1] += tanhf(acc1[v] + mu_b[p * DS_ + d1]);
    mu[r + d2] += tanhf(acc2[v] + mu_b[p * DS_ + d2]);
    mu[r + d3] += tanhf(acc3[v] + mu_b[p * DS_ + d3]);
  }
}

// ---------------------------------------------------------------- FFN layer 1
__global__ void __launch_bounds__(256) ffn1_kernel(
    const float* __restrict__ features, const float* __restrict__ meaning,
    const float* __restrict__ stats, const bf16_t* __restrict__ W1bf,
    const float* __restrict__ ffn_b1, const float* __restrict__ ln_g,
    const float* __restrict__ ln_b, bf16_t* __restrict__ h1, int p) {
  __shared__ __align__(16) bf16_t panel[128 * LDST];
  int t = threadIdx.x;
  int tok0 = blockIdx.x * 128;
  const float* lg = ln_g + p * CDIM;
  const float* lb = ln_b + p * CDIM;

  // Stage hn = LayerNorm(combined) as bf16 pairs.
  for (int i = t; i < 128 * (KPAD / 2); i += 256) {
    int r = i / (KPAD / 2);
    int c = (i - r * (KPAD / 2)) * 2;
    int tok = tok0 + r;
    unsigned pk;
    if (c >= CDIM) {
      pk = 0u;
    } else {
      float2 x = (c < DF_) ? *(const float2*)(features + (long)tok * DF_ + c)
                           : *(const float2*)(meaning + (long)(tok >> 9) * DF_ + c - DF_);
      float mean = stats[(long)tok * 2], rstd = stats[(long)tok * 2 + 1];
      float h0 = (x.x - mean) * rstd * lg[c] + lb[c];
      float h1v = (x.y - mean) * rstd * lg[c + 1] + lb[c + 1];
      pk = (unsigned)f2bf_bits(h0) | ((unsigned)f2bf_bits(h1v) << 16);
    }
    ((unsigned*)panel)[(r * LDST + c) >> 1] = pk;
  }
  __syncthreads();

  const bf16_t* W = W1bf + (long)p * KPAD * KPAD;
  int wave = t >> 5, lane = t & 31;
  int col = lane & 15;
  int kh = (lane < 16) ? 0 : 16;
  int ksel = (lane < 16) ? 0 : 16;
  const bf16_t* abase = panel + (wave * 16 + (lane & 15)) * LDST;
  int rowoff = (lane < 16) ? 0 : 8;
  int tokenM = tok0 + wave * 16;

  for (int n = 0; n < KPAD / 32; ++n) {          // 2 N-tiles per iteration
    int h0c = (2 * n) * 16 + col;
    int h1c = h0c + 16;
    const bf16_t* Wc0 = W + (long)h0c * KPAD + kh;
    const bf16_t* Wc1 = W + (long)h1c * KPAD + kh;
    v8f accA = {}, accB = {};
#pragma unroll 1
    for (int k0 = 0; k0 < KPAD; k0 += 32) {
      v16bf a  = load_bf16x16(abase + k0 + ksel, abase + k0 + ksel + 8);
      v16bf b0 = load_bf16x16(Wc0 + k0, Wc0 + k0 + 8);
      v16bf b1 = load_bf16x16(Wc1 + k0, Wc1 + k0 + 8);
      accA = wmma_bf16(a, b0, accA);
      accB = wmma_bf16(a, b1, accB);
    }
    float bbA = (h0c < CDIM) ? ffn_b1[p * CDIM + h0c] : 0.f;
    float bbB = (h1c < CDIM) ? ffn_b1[p * CDIM + h1c] : 0.f;
#pragma unroll
    for (int v = 0; v < 8; ++v) {
      long row = (long)(tokenM + rowoff + v) * KPAD;
      float xa = accA[v] + bbA;
      float xb = accB[v] + bbB;
      h1[row + h0c] = f2bf(0.5f * xa * (1.f + erff(xa * 0.70710678118654752f)));
      h1[row + h1c] = f2bf(0.5f * xb * (1.f + erff(xb * 0.70710678118654752f)));
    }
  }
}

// ---------------------------------------------------------------- FFN layer 2
__global__ void __launch_bounds__(256) ffn2_kernel(
    const bf16_t* __restrict__ h1, const bf16_t* __restrict__ W2bf,
    const float* __restrict__ ffn_b2, float* __restrict__ features, int p) {
  __shared__ __align__(16) bf16_t panel[128 * LDST];
  int t = threadIdx.x;
  int tok0 = blockIdx.x * 128;

  // Stage 128 rows of h1 (already bf16) via the CDNA5 async global->LDS path:
  // no VGPR round-trip, tracked with ASYNCcnt.
  {
    const uint4* src = (const uint4*)(h1 + (long)tok0 * KPAD);  // contiguous panel
    const int ROWQ = KPAD / 8;   // 76 uint4 per source row
    const int LDSQ = LDST / 8;   // 77 uint4 per LDS row
    for (int i = t; i < 128 * ROWQ; i += 256) {
      int r = i / ROWQ, c = i - r * ROWQ;
      unsigned lds_off = (unsigned)(uintptr_t)((uint4*)panel + (r * LDSQ + c));
      unsigned long long gaddr = (unsigned long long)(uintptr_t)(src + i);
      asm volatile("global_load_async_to_lds_b128 %0, %1, off"
                   :: "v"(lds_off), "v"(gaddr) : "memory");
    }
    asm volatile("s_wait_asynccnt 0" ::: "memory");
  }
  __syncthreads();

  int wave = t >> 5, lane = t & 31;
  int col = lane & 15;
  int kh = (lane < 16) ? 0 : 16;
  int ksel = (lane < 16) ? 0 : 16;
  const bf16_t* abase = panel + (wave * 16 + (lane & 15)) * LDST;
  const bf16_t* W = W2bf + (long)p * N2P * KPAD;
  int rowoff = (lane < 16) ? 0 : 8;
  int tokenM = tok0 + wave * 16;

  for (int n = 0; n < N2P / 32; ++n) {           // 2 N-tiles per iteration
    int f0 = (2 * n) * 16 + col;
    int f1 = f0 + 16;
    const bf16_t* Wc0 = W + (long)f0 * KPAD + kh;
    const bf16_t* Wc1 = W + (long)f1 * KPAD + kh;
    v8f accA = {}, accB = {};
#pragma unroll 1
    for (int k0 = 0; k0 < KPAD; k0 += 32) {
      v16bf a  = load_bf16x16(abase + k0 + ksel, abase + k0 + ksel + 8);
      v16bf b0 = load_bf16x16(Wc0 + k0, Wc0 + k0 + 8);
      v16bf b1 = load_bf16x16(Wc1 + k0, Wc1 + k0 + 8);
      accA = wmma_bf16(a, b0, accA);
      accB = wmma_bf16(a, b1, accB);
    }
    if (f0 < DF_) {
      float bb = ffn_b2[p * DF_ + f0];
#pragma unroll
      for (int v = 0; v < 8; ++v)
        features[(long)(tokenM + rowoff + v) * DF_ + f0] += accA[v] + bb;
    }
    if (f1 < DF_) {
      float bb = ffn_b2[p * DF_ + f1];
#pragma unroll
      for (int v = 0; v < 8; ++v)
        features[(long)(tokenM + rowoff + v) * DF_ + f1] += accB[v] + bb;
    }
  }
}

// ---------------------------------------------------------------- launch
extern "C" void kernel_launch(void* const* d_in, const int* in_sizes, int n_in,
                              void* d_out, int out_size, void* d_ws, size_t ws_size,
                              hipStream_t stream) {
  const int*           token_ids = (const int*)d_in[0];
  const unsigned char* mask      = (const unsigned char*)d_in[1];  // jax bool = 1 byte
  const float* mu_table     = (const float*)d_in[2];
  const float* logvar_table = (const float*)d_in[3];
  const float* alpha_table  = (const float*)d_in[4];
  const float* feat_table   = (const float*)d_in[5];
  const float* log_tau      = (const float*)d_in[6];
  const float* pos_mu       = (const float*)d_in[7];
  const float* pos_alpha    = (const float*)d_in[8];
  const float* mu_W   = (const float*)d_in[9];
  const float* mu_b   = (const float*)d_in[10];
  const float* gate_W = (const float*)d_in[11];
  const float* gate_b = (const float*)d_in[12];
  const float* ln_g   = (const float*)d_in[13];
  const float* ln_b   = (const float*)d_in[14];
  const float* ffn_W1 = (const float*)d_in[15];
  const float* ffn_b1 = (const float*)d_in[16];
  const float* ffn_W2 = (const float*)d_in[17];
  const float* ffn_b2 = (const float*)d_in[18];

  char* ws = (char*)d_ws;
  float* mu       = (float*)ws; ws += (size_t)NTOK * DS_ * 4;
  float* inv_var  = (float*)ws; ws += (size_t)NTOK * DS_ * 4;
  float* alpha    = (float*)ws; ws += (size_t)NTOK * 4;
  float* features = (float*)ws; ws += (size_t)NTOK * DF_ * 4;
  float* meaning  = (float*)ws; ws += (size_t)B_ * DF_ * 4;
  float* stats    = (float*)ws; ws += (size_t)NTOK * 2 * 4;
  bf16_t* h1      = (bf16_t*)ws; ws += (size_t)NTOK * KPAD * 2;
  bf16_t* W1bf    = (bf16_t*)ws; ws += (size_t)3 * KPAD * KPAD * 2;
  bf16_t* W2bf    = (bf16_t*)ws; ws += (size_t)3 * N2P * KPAD * 2;
  bf16_t* muWbf   = (bf16_t*)ws; ws += (size_t)3 * DS_ * KPAD * 2;

  convert_w_kernel<<<2048, 256, 0, stream>>>(ffn_W1, W1bf, 3, KPAD, KPAD, CDIM, CDIM);
  convert_w_kernel<<<1024, 256, 0, stream>>>(ffn_W2, W2bf, 3, N2P, KPAD, DF_, CDIM);
  convert_w_kernel<<<256, 256, 0, stream>>>(mu_W, muWbf, 3, DS_, KPAD, DS_, CDIM);

  prep_kernel<<<NTOK, 256, 0, stream>>>(token_ids, mask, mu_table, logvar_table,
                                        alpha_table, feat_table, pos_mu, pos_alpha,
                                        mu, inv_var, alpha, features);

  for (int p = 0; p < 4; ++p) {
    float* mout = (p == 3) ? (float*)d_out : meaning;
    render_kernel<<<B_, 256, 0, stream>>>(mu, inv_var, alpha, features, mask, log_tau, mout);
    if (p == 3) break;
    stats_kernel<<<NTOK / 8, 256, 0, stream>>>(features, meaning, gate_W, gate_b, mask,
                                               alpha, stats, p);
    mu_gemm_kernel<<<NTOK / 128, 256, 0, stream>>>(features, meaning, muWbf, mu_b, mu, p);
    ffn1_kernel<<<NTOK / 128, 256, 0, stream>>>(features, meaning, stats, W1bf, ffn_b1,
                                                ln_g, ln_b, h1, p);
    ffn2_kernel<<<NTOK / 128, 256, 0, stream>>>(h1, W2bf, ffn_b2, features, p);
  }
}
